// convDU3_44470091383399
// MI455X (gfx1250) — matrix operand order repaired and checked by reading, not compiled, not run on hardware
//
#include <hip/hip_runtime.h>

// ---------------------------------------------------------------------------
// Center-tap 9x1 conv == GEMM:  out[n,o,p] = relu( sum_c fea[n,c,p]*Wc[o,c] + b[o] )
// M = 512 (c_out), N = 16384 (n*h*w), K = 2048 (c_in)
// f32 -> f16 operands, f32 accumulation via V_WMMA_F32_16X16X32_F16 (wave32).
// Double-buffered LDS; A tile staged with global_load_async_to_lds_b128.
// ---------------------------------------------------------------------------

typedef __attribute__((ext_vector_type(16))) _Float16 v16h;
typedef __attribute__((ext_vector_type(8)))  _Float16 v8h;
typedef __attribute__((ext_vector_type(8)))  float    v8f;

#define CIN        2048
#define COUT       512
#define NPIX       1024   // 32*32 pixels per image
#define TAPS       9
#define CENTER_TAP 4

#define KSTEP  64
#define NSLAB  (CIN / KSTEP)   // 32
#define LDAH   72              // A-tile LDS row stride in halves (144 B rows, 16B-aligned)
#define LDBH   72              // B-tile LDS row stride in halves

// --- Kernel 1: gather center tap, convert f32 -> f16, contiguous [o][c] -----
__global__ __launch_bounds__(256)
void repack_wc_f16(const float* __restrict__ W, _Float16* __restrict__ WcH) {
  int i = blockIdx.x * 256 + threadIdx.x;      // i = o*CIN + c
  if (i < COUT * CIN)
    WcH[i] = (_Float16)W[(size_t)i * TAPS + CENTER_TAP];
}

// --- Kernel 2: tiled WMMA GEMM ---------------------------------------------
// Workgroup tile: 64 M x 256 N. 8 waves, each wave: 32 M x 64 N (8x v8f acc).
__global__ __launch_bounds__(256)
void conv_center_wmma(const float* __restrict__ fea,
                      const _Float16* __restrict__ WcH,
                      const float* __restrict__ bias,
                      float* __restrict__ out) {
  __shared__ _Float16 Alds[2][64 * LDAH];     // 2 x  9216 B : 64 o  x 64 k
  __shared__ _Float16 Blds[2][256 * LDBH];    // 2 x 36864 B : 256 p x 64 k (transposed)

  const int t      = threadIdx.x;
  const int lane   = t & 31;
  const int lnlo   = lane & 15;               // N / M index within 16
  const int lnhi   = lane >> 4;               // K-half (A/B), M-half (C/D)
  const int wave   = t >> 5;                  // 0..7
  const int wave_m = wave & 1;                // 2 waves along M (32 rows each)
  const int wave_n = wave >> 1;               // 4 waves along N (64 cols each)

  const int wg_m   = blockIdx.x;              // 8 tiles of 64 outputs (fast dim:
  const int wg_n   = blockIdx.y;              //   adjacent blocks share B tile in L2)
  const int om0    = wg_m * 64;
  const int n_img  = wg_n >> 2;               // 256-pixel tile lies inside one image
  const int p_base = (wg_n & 3) * 256;

  // this thread's pixel column for B staging (stride NPIX floats between k's)
  const float* feaT = fea + (size_t)n_img * CIN * NPIX + p_base + t;

  const v8f vzero = {};
  v8f acc[2][4];
#pragma unroll
  for (int mi = 0; mi < 2; ++mi)
#pragma unroll
    for (int ni = 0; ni < 4; ++ni)
      acc[mi][ni] = vzero;

  // ---- staging helpers -----------------------------------------------------
  // A tile: pure f16 copy -> use CDNA5 async global->LDS DMA (ASYNCcnt).
  auto stageA_async = [&](int buf, int k0) {
#pragma unroll
    for (int j = 0; j < 2; ++j) {
      int f   = t + j * 256;                  // 512 chunks of 8 halves
      int row = f >> 3;
      int c8  = (f & 7) * 8;
      // Low 32 bits of a generic LDS pointer are the LDS byte offset (ISA 10.2).
      unsigned ldsoff = (unsigned)(unsigned long long)&Alds[buf][row * LDAH + c8];
      unsigned long long gaddr =
          (unsigned long long)(WcH + (size_t)(om0 + row) * CIN + k0 + c8);
      asm volatile("global_load_async_to_lds_b128 %0, %1, off"
                   :: "v"(ldsoff), "v"(gaddr) : "memory");
    }
  };

  float regB[64];                             // next B slab, raw f32, in flight
  auto loadB = [&](int k0) {
#pragma unroll
    for (int u = 0; u < 64; ++u)
      regB[u] = feaT[(size_t)(k0 + u) * NPIX];
  };
  auto storeB = [&](int buf) {                // f32 -> f16, transposed [p][k]
#pragma unroll
    for (int j = 0; j < 8; ++j) {
      v8h tmp;
#pragma unroll
      for (int u = 0; u < 8; ++u)
        tmp[u] = (_Float16)regB[j * 8 + u];
      *(v8h*)&Blds[buf][t * LDBH + j * 8] = tmp;   // 16B-aligned ds_store_b128
    }
  };

  auto compute = [&](int buf) {
#pragma unroll
    for (int kk = 0; kk < KSTEP; kk += 32) {
      // A frags: halves 0..7 = K(lnhi*8..+7), 8..15 = K(16+lnhi*8..+7) (ISA 7.12.2)
      v16h a[2];
#pragma unroll
      for (int mi = 0; mi < 2; ++mi) {
        const _Float16* ap =
            &Alds[buf][(wave_m * 32 + mi * 16 + lnlo) * LDAH + kk + lnhi * 8];
        v8h lo = *(const v8h*)ap;
        v8h hi = *(const v8h*)(ap + 16);
        a[mi] = __builtin_shufflevector(lo, hi,
                 0,1,2,3,4,5,6,7,8,9,10,11,12,13,14,15);
      }
      // B frags: halves 0..7 = K(lnhi*8..+7), 8..15 = K(16+lnhi*8..+7),
      // matching the A fragment order in the transposed LDS tile.
#pragma unroll
      for (int ni = 0; ni < 4; ++ni) {
        const _Float16* bp =
            &Blds[buf][(wave_n * 64 + ni * 16 + lnlo) * LDBH + kk + lnhi * 8];
        v8h lo = *(const v8h*)bp;
        v8h hi = *(const v8h*)(bp + 16);
        v16h b = __builtin_shufflevector(lo, hi,
                  0,1,2,3,4,5,6,7,8,9,10,11,12,13,14,15);
#pragma unroll
        for (int mi = 0; mi < 2; ++mi)
          acc[mi][ni] = __builtin_amdgcn_wmma_f32_16x16x32_f16(
              /*neg_a=*/false, a[mi], /*neg_b=*/false, b,
              /*c_mod=*/(short)0, acc[mi][ni],
              /*reuse_a=*/false, /*reuse_b=*/false);
      }
    }
  };

  // ---- double-buffered main loop ------------------------------------------
  stageA_async(0, 0);
  loadB(0);
  storeB(0);
  asm volatile("s_wait_asynccnt 0" ::: "memory");
  __syncthreads();

  int cur = 0;
#pragma nounroll
  for (int it = 0; it < NSLAB; ++it) {
    const bool has_next = (it + 1) < NSLAB;
    const int  k0n      = (it + 1) * KSTEP;
    if (has_next) {
      stageA_async(cur ^ 1, k0n);   // async DMA into other buffer
      loadB(k0n);                   // global loads in flight during compute
    }
    compute(cur);                   // ds_load_b128 + v_wmma on current buffer
    if (has_next) {
      storeB(cur ^ 1);              // waits LOADcnt, converts, ds_store
      asm volatile("s_wait_asynccnt 0" ::: "memory");
    }
    __syncthreads();
    cur ^= 1;
  }

  // ---- Epilogue: bias + ReLU. D layout: VGPR r -> M = r + 8*lnhi, N = lnlo.
#pragma unroll
  for (int mi = 0; mi < 2; ++mi) {
#pragma unroll
    for (int r = 0; r < 8; ++r) {
      int o = om0 + wave_m * 32 + mi * 16 + lnhi * 8 + r;
      float bv = bias[o];
      size_t obase = ((size_t)n_img * COUT + o) * NPIX + p_base;
#pragma unroll
      for (int ni = 0; ni < 4; ++ni) {
        int p = wave_n * 64 + ni * 16 + lnlo;
        float v = acc[mi][ni][r] + bv;
        out[obase + p] = v > 0.f ? v : 0.f;
      }
    }
  }
}

// ---------------------------------------------------------------------------
extern "C" void kernel_launch(void* const* d_in, const int* in_sizes, int n_in,
                              void* d_out, int out_size, void* d_ws, size_t ws_size,
                              hipStream_t stream) {
  (void)in_sizes; (void)n_in; (void)out_size; (void)ws_size;
  const float* fea  = (const float*)d_in[0];   // (16, 2048, 32, 32) f32
  const float* W    = (const float*)d_in[1];   // (512, 2048, 9, 1)  f32
  const float* bias = (const float*)d_in[2];   // (512,)             f32
  float*    out = (float*)d_out;               // (16, 512, 32, 32)  f32
  _Float16* WcH = (_Float16*)d_ws;             // 512*2048*2 = 2 MiB scratch

  repack_wc_f16<<<(COUT * CIN + 255) / 256, 256, 0, stream>>>(W, WcH);
  // grid.x = M tiles so the 8 blocks sharing one fea B-tile are dispatch-adjacent
  conv_center_wmma<<<dim3(8, 64), 256, 0, stream>>>(fea, WcH, bias, out);
}
